// SwinTransformerV2Block_55113020343071
// MI455X (gfx1250) — compile-verified, matrix operand order checked by reading, not run on hardware
//
#include <hip/hip_runtime.h>
#include <hip/hip_bf16.h>
#include <math.h>

typedef _Float16 half_t;
typedef half_t v16h __attribute__((ext_vector_type(16)));
typedef half_t v8h  __attribute__((ext_vector_type(8)));
typedef float  v8f  __attribute__((ext_vector_type(8)));

#define DIMC   256
#define HEADS  8
#define HD     32
#define WSZ    8
#define SHIFTS 4
#define MTOK   64
#define IMG    256
#define HIDDEN 1024
#define LDA    264   // padded f16 row stride (264 halves = 528B, bank-spread, 16B-aligned)

// workspace layout (in halves): qkvT 768x256 | projT 256x256 | fc1T 1024x256 | fc2T 256x1024
#define OFF_QKVT 0
#define OFF_PROJT (768*256)
#define OFF_FC1T  (OFF_PROJT + 256*256)
#define OFF_FC2T  (OFF_FC1T + 1024*256)
#define TOT_H     (OFF_FC2T + 256*1024)
#define OFF_BIAS_BYTES ((size_t)TOT_H * 2)   // then float bias16[8][64][64]

// ---- WMMA helpers -----------------------------------------------------------

// 16x32 f16 fragment (A-style; B comes from pre-transposed N x K weights).
// CDNA5 layout: lanes 0-15 row M=lane hold K 0..7 then 16..23;
// lanes 16-31 hold K 8..15 then 24..31.
__device__ __forceinline__ v16h frag_ld(const half_t* p0, int ld, int lane) {
  const half_t* p = p0 + (lane & 15) * ld + ((lane >> 4) << 3);
  v8h lo = *(const v8h*)p;
  v8h hi = *(const v8h*)(p + 16);
  return __builtin_shufflevector(lo, hi, 0,1,2,3,4,5,6,7,8,9,10,11,12,13,14,15);
}

__device__ __forceinline__ v8f wmma_f16(v16h a, v16h b, v8f c) {
  return __builtin_amdgcn_wmma_f32_16x16x32_f16(false, a, false, b, (short)0, c,
                                                false, false);
}

#define ZERO8 ((v8f){0.f,0.f,0.f,0.f,0.f,0.f,0.f,0.f})

// async global -> LDS copy of 16 bytes (per-lane), tracked on ASYNCcnt
__device__ __forceinline__ void async_b128(unsigned lds_addr, unsigned long long gaddr) {
  asm volatile("global_load_async_to_lds_b128 %0, %1, off"
               :: "v"(lds_addr), "v"(gaddr) : "memory");
}
__device__ __forceinline__ void wait_async0() {
  asm volatile("s_wait_asynccnt 0x0" ::: "memory");
}

// ---- Prep kernel 1: transpose + f32->f16 convert all weights ---------------

__global__ __launch_bounds__(256) void prep_weights(
    const float* __restrict__ qkv_w, const float* __restrict__ proj_w,
    const float* __restrict__ fc1_w, const float* __restrict__ fc2_w,
    half_t* __restrict__ wh)
{
  int idx = blockIdx.x * 256 + threadIdx.x;
  if (idx < OFF_PROJT) {                       // qkvT[n][k] = qkv_w[k][n]
    int n = idx >> 8, k = idx & 255;
    wh[idx] = (half_t)qkv_w[k * 768 + n];
  } else if (idx < OFF_FC1T) {                 // projT
    int t = idx - OFF_PROJT; int n = t >> 8, k = t & 255;
    wh[idx] = (half_t)proj_w[k * 256 + n];
  } else if (idx < OFF_FC2T) {                 // fc1T (1024 x 256)
    int t = idx - OFF_FC1T; int n = t >> 8, k = t & 255;
    wh[idx] = (half_t)fc1_w[k * 1024 + n];
  } else if (idx < TOT_H) {                    // fc2T (256 x 1024)
    int t = idx - OFF_FC2T; int n = t >> 10, k = t & 1023;
    wh[idx] = (half_t)fc2_w[k * 256 + n];
  }
}

// ---- Prep kernel 2: CPB MLP -> 16*sigmoid bias table -----------------------

__device__ __forceinline__ float logrel(int i) {
  float v = (float)(i - 7) * (8.f / 7.f);
  float a = log2f(fabsf(v) + 1.f) * (1.f / 3.f);   // /log2(8)
  return copysignf(a, v);
}

__global__ __launch_bounds__(256) void prep_bias(
    const float* __restrict__ w1, const float* __restrict__ b1,
    const float* __restrict__ w2, float* __restrict__ bias16)
{
  __shared__ float bt[225 * 8];
  int tid = threadIdx.x;
  if (tid < 225) {
    int a = tid / 15, bcol = tid % 15;
    float fx = logrel(a), fy = logrel(bcol);
    float acc[8];
#pragma unroll
    for (int h = 0; h < 8; ++h) acc[h] = 0.f;
    for (int j = 0; j < 512; ++j) {
      float hv = fmaxf(fx * w1[j] + fy * w1[512 + j] + b1[j], 0.f);
#pragma unroll
      for (int h = 0; h < 8; ++h) acc[h] += hv * w2[j * 8 + h];
    }
#pragma unroll
    for (int h = 0; h < 8; ++h) bt[tid * 8 + h] = acc[h];
  }
  __syncthreads();
  for (int q = tid; q < 4096; q += 256) {
    int i = q >> 6, j = q & 63;
    int dy = (i >> 3) - (j >> 3) + 7;
    int dx = (i & 7) - (j & 7) + 7;
    int t = dy * 15 + dx;
#pragma unroll
    for (int h = 0; h < 8; ++h)
      bias16[h * 4096 + q] = 16.f / (1.f + __expf(-bt[t * 8 + h]));
  }
}

// ---- Main fused Swin-V2 block kernel: one 8x8 window per workgroup ---------

__global__ __launch_bounds__(256) void swin_block(
    const float* __restrict__ x,
    const float* __restrict__ logit_scale,
    const float* __restrict__ proj_b,
    const float* __restrict__ ln1_w, const float* __restrict__ ln1_b,
    const float* __restrict__ fc1_b, const float* __restrict__ fc2_b,
    const float* __restrict__ ln2_w, const float* __restrict__ ln2_b,
    const half_t* __restrict__ wh, const float* __restrict__ bias16,
    float* __restrict__ out)
{
  // attention scratch overlaid with the raw f32 input staging buffer:
  // xstage is only live before the head loop starts.
  union SMem {
    float xstage[MTOK * DIMC];                 // 64 KB raw window (f32)
    struct {
      half_t S16[MTOK * LDA];                  // o / gelu chunk / mlp staging
      half_t qb[MTOK * HD];
      half_t kb[MTOK * HD];
      half_t vT[HD * MTOK];                    // v transposed (N=d x K=j)
      half_t p16[MTOK * MTOK];
      float  lg[MTOK * MTOK];
    } at;
  };
  __shared__ SMem U;
  __shared__ half_t XY16[MTOK * LDA];          // x window -> y (in place)
  __shared__ float  muA[MTOK], rsA[MTOK];
  __shared__ int    lab[MTOK];

  half_t* S16 = U.at.S16;
  half_t* qb  = U.at.qb;
  half_t* kb  = U.at.kb;
  half_t* vT  = U.at.vT;
  half_t* p16 = U.at.p16;
  float*  lg  = U.at.lg;

  const int tid  = threadIdx.x;
  const int wv   = tid >> 5;
  const int lane = tid & 31;
  const int blk  = blockIdx.x;
  const int b    = blk >> 10;
  const int wi   = blk & 1023;
  const int wy   = wi >> 5, wx = wi & 31;

  const half_t* qkvT  = wh + OFF_QKVT;
  const half_t* projT = wh + OFF_PROJT;
  const half_t* fc1T  = wh + OFF_FC1T;
  const half_t* fc2T  = wh + OFF_FC2T;

  // ---- async DMA the shifted window (f32) into LDS staging ----
  // 4096 x 16B copies: job = (c, ty, 4px-group); each group is contiguous in
  // memory even across the cyclic-shift wrap (group start gx <= 252 always).
#pragma unroll
  for (int it = 0; it < 16; ++it) {
    int job = it * 256 + tid;
    int c   = job >> 4;
    int rem = job & 15;
    int ty  = rem >> 1;
    int grp = rem & 1;
    int gy  = ((wy << 3) + ty + SHIFTS) & 255;
    int gx  = ((wx << 3) + (grp << 2) + SHIFTS) & 255;
    const float* gp = x + (((size_t)(b * DIMC + c)) << 16) + (gy << 8) + gx;
    unsigned lds = (unsigned)(unsigned long long)(uintptr_t)
                   (&U.xstage[c * MTOK + ty * 8 + grp * 4]);
    async_b128(lds, (unsigned long long)(uintptr_t)gp);
  }
  if (tid < 64) {
    int r = tid;
    int sy = (wy << 3) + (r >> 3), sx = (wx << 3) + (r & 7);
    int ry = (sy < IMG - WSZ) ? 0 : (sy < IMG - SHIFTS ? 1 : 2);
    int rx = (sx < IMG - WSZ) ? 0 : (sx < IMG - SHIFTS ? 1 : 2);
    lab[r] = ry * 3 + rx;
  }
  if (tid == 0 && blk + 1 < (int)gridDim.x)
    __builtin_prefetch(x + ((size_t)((blk + 1) >> 10) << 24), 0, 1);
  wait_async0();
  __syncthreads();

  // convert staged f32 -> f16 A-operand layout (token-major, padded rows)
  for (int it = 0; it < 64; ++it) {
    int idx = it * 256 + tid;
    int r = idx & 63, c = idx >> 6;
    XY16[r * LDA + c] = (half_t)U.xstage[c * MTOK + r];
  }
  __syncthreads();

  // ---- attention, one head at a time ----
  for (int h = 0; h < HEADS; ++h) {
    // QKV GEMM: 24 (16x16) output tiles, 3 per wave, K = 256
#pragma unroll
    for (int jj = 0; jj < 3; ++jj) {
      int job = wv + (jj << 3);
      int mat = job >> 3, tix = job & 7;
      int mi = tix & 3, ni = tix >> 2;
      const half_t* Bb = qkvT + (size_t)(mat * DIMC + h * HD + ni * 16) * DIMC;
      const half_t* Ab = XY16 + (mi * 16) * LDA;
      v8f acc = ZERO8;
#pragma unroll
      for (int k0 = 0; k0 < DIMC; k0 += 32)
        acc = wmma_f16(frag_ld(Ab + k0, LDA, lane), frag_ld(Bb + k0, DIMC, lane), acc);
      int Mo = mi * 16 + ((lane >> 4) << 3);
      int No = ni * 16 + (lane & 15);
      if (mat == 0) {
#pragma unroll
        for (int g = 0; g < 8; ++g) qb[(Mo + g) * HD + No] = (half_t)acc[g];
      } else if (mat == 1) {
#pragma unroll
        for (int g = 0; g < 8; ++g) kb[(Mo + g) * HD + No] = (half_t)acc[g];
      } else {
#pragma unroll
        for (int g = 0; g < 8; ++g) vT[No * MTOK + (Mo + g)] = (half_t)acc[g];
      }
    }
    __syncthreads();

    // cosine-attention row normalization; fold logit scale into q
    if (tid < 128) {
      int r = tid & 63, isk = tid >> 6;
      half_t* buf = isk ? kb : qb;
      float ss = 0.f;
      for (int d = 0; d < HD; ++d) { float v = (float)buf[r * HD + d]; ss += v * v; }
      float inv = 1.f / fmaxf(sqrtf(ss), 1e-12f);
      if (!isk) inv *= __expf(fminf(logit_scale[h], 4.6051702f));
      for (int d = 0; d < HD; ++d)
        buf[r * HD + d] = (half_t)((float)buf[r * HD + d] * inv);
    }
    __syncthreads();

    // logits = qn @ kn^T (K=32), + bias + shift mask
#pragma unroll
    for (int jj = 0; jj < 2; ++jj) {
      int tix = (wv << 1) + jj;
      int mi = tix >> 2, ni = tix & 3;
      v8f acc = wmma_f16(frag_ld(qb + mi * 16 * HD, HD, lane),
                         frag_ld(kb + ni * 16 * HD, HD, lane), ZERO8);
      int Mo = mi * 16 + ((lane >> 4) << 3);
      int No = ni * 16 + (lane & 15);
      const float* bh = bias16 + h * 4096;
#pragma unroll
      for (int g = 0; g < 8; ++g) {
        int i = Mo + g;
        float msk = (lab[i] == lab[No]) ? 0.f : -100.f;
        lg[i * 64 + No] = acc[g] + bh[i * 64 + No] + msk;
      }
    }
    __syncthreads();

    // softmax over j (row per thread)
    if (tid < 64) {
      float mx = -1e30f;
      for (int j = 0; j < 64; ++j) mx = fmaxf(mx, lg[tid * 64 + j]);
      float s = 0.f;
      for (int j = 0; j < 64; ++j) s += __expf(lg[tid * 64 + j] - mx);
      float invs = 1.f / s;
      for (int j = 0; j < 64; ++j)
        p16[tid * 64 + j] = (half_t)(__expf(lg[tid * 64 + j] - mx) * invs);
    }
    __syncthreads();

    // o_h = p @ v (K=64), one tile per wave, into S16 head columns
    {
      int mi = wv & 3, ni = wv >> 2;
      v8f acc = ZERO8;
#pragma unroll
      for (int k0 = 0; k0 < 64; k0 += 32)
        acc = wmma_f16(frag_ld(p16 + mi * 16 * 64 + k0, 64, lane),
                       frag_ld(vT + ni * 16 * 64 + k0, 64, lane), acc);
      int Mo = mi * 16 + ((lane >> 4) << 3);
      int No = h * HD + ni * 16 + (lane & 15);
#pragma unroll
      for (int g = 0; g < 8; ++g) S16[(Mo + g) * LDA + No] = (half_t)acc[g];
    }
    __syncthreads();
  }

  // ---- output projection: 64 tiles, 8 per wave, K = 256 ----
  v8f pres[8];
#pragma unroll
  for (int t = 0; t < 8; ++t) {
    int tix = (wv << 3) + t;
    int mi = tix >> 4, ni = tix & 15;
    v8f acc = ZERO8;
#pragma unroll
    for (int k0 = 0; k0 < DIMC; k0 += 32)
      acc = wmma_f16(frag_ld(S16 + mi * 16 * LDA + k0, LDA, lane),
                     frag_ld(projT + (size_t)(ni * 16) * DIMC + k0, DIMC, lane), acc);
    pres[t] = acc;
  }
  __syncthreads();
#pragma unroll
  for (int t = 0; t < 8; ++t) {
    int tix = (wv << 3) + t;
    int mi = tix >> 4, ni = tix & 15;
    int Mo = mi * 16 + ((lane >> 4) << 3);
    int No = ni * 16 + (lane & 15);
    float pb = proj_b[No];
#pragma unroll
    for (int g = 0; g < 8; ++g) S16[(Mo + g) * LDA + No] = (half_t)(pres[t][g] + pb);
  }
  __syncthreads();

  // ---- y = LN1(o) + x  (in place into XY16) ----
  if (tid < 64) {
    float mu = 0.f;
    for (int c = 0; c < DIMC; ++c) mu += (float)S16[tid * LDA + c];
    mu *= (1.f / DIMC);
    float var = 0.f;
    for (int c = 0; c < DIMC; ++c) {
      float d = (float)S16[tid * LDA + c] - mu; var += d * d;
    }
    float rs = rsqrtf(var * (1.f / DIMC) + 1e-5f);
    for (int c = 0; c < DIMC; ++c) {
      float yv = ((float)S16[tid * LDA + c] - mu) * rs * ln1_w[c] + ln1_b[c]
               + (float)XY16[tid * LDA + c];
      XY16[tid * LDA + c] = (half_t)yv;
    }
  }
  __syncthreads();

  // ---- MLP: 4 chunks of 256 hidden; fc2 accumulates persistently ----
  v8f acc2[8];
#pragma unroll
  for (int t = 0; t < 8; ++t) acc2[t] = ZERO8;

  for (int kc = 0; kc < 4; ++kc) {
    v8f r1[8];
#pragma unroll
    for (int t = 0; t < 8; ++t) {
      int tix = (wv << 3) + t;
      int mi = tix >> 4, ni = tix & 15;
      const half_t* Bb = fc1T + (size_t)(kc * 256 + ni * 16) * DIMC;
      v8f acc = ZERO8;
#pragma unroll
      for (int k0 = 0; k0 < DIMC; k0 += 32)
        acc = wmma_f16(frag_ld(XY16 + mi * 16 * LDA + k0, LDA, lane),
                       frag_ld(Bb + k0, DIMC, lane), acc);
      r1[t] = acc;
    }
#pragma unroll
    for (int t = 0; t < 8; ++t) {
      int tix = (wv << 3) + t;
      int mi = tix >> 4, ni = tix & 15;
      int Mo = mi * 16 + ((lane >> 4) << 3);
      int No = ni * 16 + (lane & 15);
      float bb = fc1_b[kc * 256 + No];
#pragma unroll
      for (int g = 0; g < 8; ++g) {
        float v = r1[t][g] + bb;
        float ge = 0.5f * v * (1.f + erff(v * 0.70710678f));
        S16[(Mo + g) * LDA + No] = (half_t)ge;
      }
    }
    __syncthreads();
#pragma unroll
    for (int t = 0; t < 8; ++t) {
      int tix = (wv << 3) + t;
      int mi = tix >> 4, ni = tix & 15;
      const half_t* Bb = fc2T + (size_t)(ni * 16) * HIDDEN + kc * 256;
      v8f acc = acc2[t];
#pragma unroll
      for (int k0 = 0; k0 < DIMC; k0 += 32)
        acc = wmma_f16(frag_ld(S16 + mi * 16 * LDA + k0, LDA, lane),
                       frag_ld(Bb + k0, HIDDEN, lane), acc);
      acc2[t] = acc;
    }
    __syncthreads();
  }

  // mlp result into S16
#pragma unroll
  for (int t = 0; t < 8; ++t) {
    int tix = (wv << 3) + t;
    int mi = tix >> 4, ni = tix & 15;
    int Mo = mi * 16 + ((lane >> 4) << 3);
    int No = ni * 16 + (lane & 15);
    float bb = fc2_b[No];
#pragma unroll
    for (int g = 0; g < 8; ++g) S16[(Mo + g) * LDA + No] = (half_t)(acc2[t][g] + bb);
  }
  __syncthreads();

  // LN2 stats
  if (tid < 64) {
    float mu = 0.f;
    for (int c = 0; c < DIMC; ++c) mu += (float)S16[tid * LDA + c];
    mu *= (1.f / DIMC);
    float var = 0.f;
    for (int c = 0; c < DIMC; ++c) {
      float d = (float)S16[tid * LDA + c] - mu; var += d * d;
    }
    muA[tid] = mu;
    rsA[tid] = rsqrtf(var * (1.f / DIMC) + 1e-5f);
  }
  __syncthreads();

  // ---- final: out = LN2(mlp) + y, window-reverse + un-shift ----
  for (int it = 0; it < 64; ++it) {
    int c = (it << 2) + (tid >> 6);
    int r = tid & 63;
    float mv = ((float)S16[r * LDA + c] - muA[r]) * rsA[r] * ln2_w[c] + ln2_b[c];
    float yv = mv + (float)XY16[r * LDA + c];
    int gy = ((wy << 3) + (r >> 3) + SHIFTS) & 255;
    int gx = ((wx << 3) + (r & 7) + SHIFTS) & 255;
    out[(((size_t)(b * DIMC + c)) << 16) + (gy << 8) + gx] = yv;
  }
}

// ---- launch ----------------------------------------------------------------

extern "C" void kernel_launch(void* const* d_in, const int* in_sizes, int n_in,
                              void* d_out, int out_size, void* d_ws, size_t ws_size,
                              hipStream_t stream) {
  (void)in_sizes; (void)n_in; (void)out_size; (void)ws_size;
  const float* x        = (const float*)d_in[0];
  const float* qkv_w    = (const float*)d_in[1];
  const float* lscale   = (const float*)d_in[2];
  const float* proj_w   = (const float*)d_in[3];
  const float* proj_b   = (const float*)d_in[4];
  const float* ln1_w    = (const float*)d_in[5];
  const float* ln1_b    = (const float*)d_in[6];
  const float* fc1_w    = (const float*)d_in[7];
  const float* fc1_b    = (const float*)d_in[8];
  const float* fc2_w    = (const float*)d_in[9];
  const float* fc2_b    = (const float*)d_in[10];
  const float* ln2_w    = (const float*)d_in[11];
  const float* ln2_b    = (const float*)d_in[12];
  const float* cpb_w1   = (const float*)d_in[13];
  const float* cpb_b1   = (const float*)d_in[14];
  const float* cpb_w2   = (const float*)d_in[15];

  half_t* wh     = (half_t*)d_ws;
  float*  bias16 = (float*)((char*)d_ws + OFF_BIAS_BYTES);
  float*  out    = (float*)d_out;

  prep_weights<<<TOT_H / 256, 256, 0, stream>>>(qkv_w, proj_w, fc1_w, fc2_w, wh);
  prep_bias<<<1, 256, 0, stream>>>(cpb_w1, cpb_b1, cpb_w2, bias16);
  swin_block<<<2048, 256, 0, stream>>>(x, lscale, proj_b, ln1_w, ln1_b,
                                       fc1_b, fc2_b, ln2_w, ln2_b,
                                       wh, bias16, out);
}
